// SelfAttention_57561151701357
// MI455X (gfx1250) — matrix operand order repaired and checked by reading, not compiled, not run on hardware
//
#include <hip/hip_runtime.h>

// ---------------------------------------------------------------------------
// Types for CDNA5 WMMA (wave32): D(16x16 f32) = A(16x32 bf16) x B(32x16 bf16) + C
// ---------------------------------------------------------------------------
typedef __attribute__((ext_vector_type(16))) __bf16 v16bf;
typedef __attribute__((ext_vector_type(8)))  float  v8f;

union BfFrag {
    v16bf v;
    unsigned int w[8];       // w[i] packs elements (2i, 2i+1): lo/hi halves of VGPR i
    unsigned short u[16];
};

__device__ __forceinline__ v8f vzero() {
    v8f z = {0.f, 0.f, 0.f, 0.f, 0.f, 0.f, 0.f, 0.f};
    return z;
}

// f32 -> bf16 (round to nearest even)
__device__ __forceinline__ unsigned short f2bf(float f) {
    union { float f; unsigned u; } x; x.f = f;
    unsigned r = x.u + 0x7FFFu + ((x.u >> 16) & 1u);
    return (unsigned short)(r >> 16);
}

// A-matrix (16x32 bf16) K-pair base for VGPR v (contiguous, matches B):
//  lanes 0-15 hold K {0..15}; lanes 16-31 hold K {16..31}
__device__ __forceinline__ int AK0(int v, int hi) {
    return 2 * v + hi * 16;
}

__device__ __forceinline__ v8f wmma_bf16(const BfFrag& a, const BfFrag& b, v8f c) {
    return __builtin_amdgcn_wmma_f32_16x16x32_bf16(
        /*neg_a=*/false, a.v, /*neg_b=*/false, b.v,
        /*c_mod=*/(short)0, c, /*reuse_a=*/false, /*reuse_b=*/false);
}

// ---- CDNA5 async global->LDS (ASYNCcnt path, bypasses VGPRs) --------------
// GV mode: vdst VGPR holds the LDS byte offset, vaddr the 64-bit global address
__device__ __forceinline__ void async_copy16(unsigned lds_byte_off, const void* gsrc) {
    asm volatile("global_load_async_to_lds_b128 %0, %1, off"
                 :: "v"(lds_byte_off), "v"(gsrc)
                 : "memory");
}
#define WAIT_ASYNC(n) asm volatile("s_wait_asynccnt %0" :: "n"(n) : "memory")

// LDS byte offset from generic pointer (ISA: LDS aperture uses addr[31:0])
__device__ __forceinline__ unsigned lds_off(const void* p) {
    return (unsigned)(unsigned long long)p;
}

// Problem dims (fixed by the reference)
#define NB   4
#define CCH  256
#define HW   4096       // 64*64
#define HEADS 4
#define DH   64
#define NQKV 768        // 3*C
#define MROWS (NB * HW) // 16384

// ---------------------------------------------------------------------------
// K0: BatchNorm statistics -> per-channel scale/shift
// ---------------------------------------------------------------------------
__global__ void __launch_bounds__(256)
bn_stats(const float* __restrict__ x, const float* __restrict__ gamma,
         const float* __restrict__ beta, float* __restrict__ scale,
         float* __restrict__ shift) {
    __shared__ float ssum[256], ssq[256];
    int ch = blockIdx.x;
    float s = 0.f, q = 0.f;
    for (int nn = 0; nn < NB; nn++) {
        const float* p = x + (size_t)(nn * CCH + ch) * HW;
        for (int i = threadIdx.x; i < HW; i += 256) {
            float v = p[i];
            s += v; q += v * v;
        }
    }
    ssum[threadIdx.x] = s; ssq[threadIdx.x] = q;
    __syncthreads();
    for (int off = 128; off > 0; off >>= 1) {
        if (threadIdx.x < off) {
            ssum[threadIdx.x] += ssum[threadIdx.x + off];
            ssq[threadIdx.x]  += ssq[threadIdx.x + off];
        }
        __syncthreads();
    }
    if (threadIdx.x == 0) {
        const float inv = 1.0f / (float)(NB * HW);
        float mean = ssum[0] * inv;
        float var  = ssq[0] * inv - mean * mean;
        float sc   = gamma[ch] * rsqrtf(var + 1e-5f);
        scale[ch] = sc;
        shift[ch] = beta[ch] - mean * sc;
    }
}

// ---------------------------------------------------------------------------
// K1: weights -> transposed bf16 (so B-fragment K-pairs are contiguous)
// ---------------------------------------------------------------------------
__global__ void __launch_bounds__(256)
prep_weights(const float* __restrict__ qkv_w, const float* __restrict__ proj_w,
             unsigned short* __restrict__ qkv_wt, unsigned short* __restrict__ proj_wt) {
    int i = blockIdx.x * 256 + threadIdx.x;
    if (i < NQKV * CCH) {
        int j = i / CCH, k = i % CCH;
        qkv_wt[i] = f2bf(qkv_w[k * NQKV + j]);
    } else {
        int t = i - NQKV * CCH;
        if (t < CCH * CCH) {
            int j = t / CCH, k = t % CCH;
            proj_wt[t] = f2bf(proj_w[k * CCH + j]);
        }
    }
}

// ---------------------------------------------------------------------------
// K2: normalized sequence (n, hw, c) in bf16
// ---------------------------------------------------------------------------
__global__ void __launch_bounds__(256)
build_seq(const float* __restrict__ x, const float* __restrict__ scale,
          const float* __restrict__ shift, unsigned short* __restrict__ seq) {
    int i = blockIdx.x * 256 + threadIdx.x;      // over NB*HW*CCH = 2^22
    int ch = i & (CCH - 1);
    int p  = (i >> 8) & (HW - 1);
    int nn = i >> 20;
    float v = x[(size_t)(nn * CCH + ch) * HW + p] * scale[ch] + shift[ch];
    seq[i] = f2bf(v);
}

// ---------------------------------------------------------------------------
// K3: QKV GEMM (16384 x 768 x 256) with epilogue scatter into scrambled-head
//     Q (nh,4096,64), K (nh,4096,64), and transposed V (nh,64,4096), all bf16.
// ---------------------------------------------------------------------------
__global__ void __launch_bounds__(256)
qkv_gemm(const unsigned short* __restrict__ seq, const unsigned short* __restrict__ wt,
         unsigned short* __restrict__ Qb, unsigned short* __restrict__ Kb,
         unsigned short* __restrict__ Vt) {
    const int lane = threadIdx.x & 31;
    const int wave = threadIdx.x >> 5;
    const int hi = (lane >> 4) & 1;
    const int ln = lane & 15;
    const int m0 = blockIdx.x * 64 + 16 * (wave & 3);
    const int c0 = blockIdx.y * 64 + 32 * (wave >> 2);

    v8f acc[2] = {vzero(), vzero()};
    for (int ks = 0; ks < 8; ks++) {
        const int kb = ks * 32;
        BfFrag a, b0, b1;
        const unsigned short* arow = seq + (size_t)(m0 + ln) * CCH + kb;
#pragma unroll
        for (int v = 0; v < 8; v++)
            a.w[v] = *(const unsigned int*)(arow + AK0(v, hi));
        const unsigned short* b0row = wt + (size_t)(c0 + ln) * CCH + kb + hi * 16;
        const unsigned short* b1row = b0row + 16 * CCH;
#pragma unroll
        for (int v = 0; v < 8; v++) {
            b0.w[v] = *(const unsigned int*)(b0row + 2 * v);
            b1.w[v] = *(const unsigned int*)(b1row + 2 * v);
        }
        acc[0] = wmma_bf16(a, b0, acc[0]);
        acc[1] = wmma_bf16(a, b1, acc[1]);
    }
    // Scatter: (row,col) of qkv -> torch-faithful head view.
#pragma unroll
    for (int sub = 0; sub < 2; sub++) {
#pragma unroll
        for (int v = 0; v < 8; v++) {
            int row = m0 + v + hi * 8;             // 0..16383
            int col = c0 + sub * 16 + ln;          // 0..767
            int nn = row >> 12, q = row & (HW - 1);
            int head = q >> 10;
            int r = ((q & 1023) << 2) + col / 192; // head-row 0..4095
            int j192 = col % 192;
            int sel = j192 >> 6, d = j192 & 63;
            unsigned short bv = f2bf(acc[sub][v]);
            int hb = nn * HEADS + head;
            if (sel == 0)       Qb[((size_t)hb * HW + r) * DH + d] = bv;
            else if (sel == 1)  Kb[((size_t)hb * HW + r) * DH + d] = bv;
            else                Vt[((size_t)hb * DH + d) * HW + r] = bv;
        }
    }
}

// ---------------------------------------------------------------------------
// K4: flash attention per (n,head). Block = 128 queries (8 waves x 16).
//     Double-buffered 64-key tiles streamed into LDS with CDNA5 async
//     global->LDS (ASYNCcnt); online softmax; fp32 WMMA accumulators.
// ---------------------------------------------------------------------------
__global__ void __launch_bounds__(256)
flash_attn(const unsigned short* __restrict__ Qb, const unsigned short* __restrict__ Kb,
           const unsigned short* __restrict__ Vt, unsigned short* __restrict__ obuf) {
    __shared__ unsigned short sK[2 * 64 * 64];   // [buf][key_local][d]
    __shared__ unsigned short sV[2 * 64 * 64];   // [buf][d][key_local]
    __shared__ unsigned short sP[8 * 16 * 64];   // per-wave P tile [row][key_local]

    const int lane = threadIdx.x & 31;
    const int wave = threadIdx.x >> 5;
    const int hi = (lane >> 4) & 1;
    const int ln = lane & 15;
    const int nh = blockIdx.y;                   // 0..15
    const int q0 = blockIdx.x * 128 + wave * 16; // query base for this wave

    const unsigned short* Qh = Qb + (size_t)nh * HW * DH;
    const unsigned short* Kh = Kb + (size_t)nh * HW * DH;
    const unsigned short* Vh = Vt + (size_t)nh * DH * HW;

    // Issue async fills of one 64-key tile pair into LDS buffer `buf`.
    // Per tile: sK = 512 x 16B chunks, sV = 512 x 16B chunks; 256 threads
    // -> 2+2 async instructions per wave per tile (ASYNCcnt += 4).
    auto issue_tile = [&](int kb, int buf) {
#pragma unroll
        for (int c = threadIdx.x; c < 512; c += 256) {
            async_copy16(lds_off(sK + buf * 4096 + c * 8),
                         Kh + (size_t)kb * 4096 + c * 8);
        }
#pragma unroll
        for (int c = threadIdx.x; c < 512; c += 256) {
            int row = c >> 3, sub = c & 7;
            async_copy16(lds_off(sV + buf * 4096 + row * 64 + sub * 8),
                         Vh + (size_t)row * HW + kb * 64 + sub * 8);
        }
    };

    // Q fragments for the two 32-wide K-steps of d
    BfFrag qf[2];
#pragma unroll
    for (int dk = 0; dk < 2; dk++) {
        const unsigned short* qrow = Qh + (size_t)(q0 + ln) * DH + dk * 32;
#pragma unroll
        for (int v = 0; v < 8; v++)
            qf[dk].w[v] = *(const unsigned int*)(qrow + AK0(v, hi));
    }

    float m[8], l[8];
    v8f o[4];
#pragma unroll
    for (int v = 0; v < 8; v++) { m[v] = -3.0e38f; l[v] = 0.f; }
#pragma unroll
    for (int t = 0; t < 4; t++) o[t] = vzero();

    issue_tile(0, 0);                    // prologue: tile 0 in flight

    for (int kb = 0; kb < HW / 64; kb++) {
        const int buf = kb & 1;
        if (kb + 1 < HW / 64) {
            issue_tile(kb + 1, buf ^ 1); // stream next tile behind compute
            WAIT_ASYNC(4);               // in-order: current tile resident
        } else {
            WAIT_ASYNC(0);
        }
        __syncthreads();                 // publish tile across waves

        const unsigned short* sKb = sK + buf * 4096;
        const unsigned short* sVb = sV + buf * 4096;

        // S = Q K^T for 4 16-key subtiles (K-dim = d = 64 -> 2 WMMA steps)
        v8f st[4];
#pragma unroll
        for (int nt = 0; nt < 4; nt++) {
            v8f accs = vzero();
#pragma unroll
            for (int dk = 0; dk < 2; dk++) {
                BfFrag bf;
                const unsigned short* brow = sKb + (nt * 16 + ln) * DH + dk * 32 + hi * 16;
#pragma unroll
                for (int v = 0; v < 8; v++)
                    bf.w[v] = *(const unsigned int*)(brow + 2 * v);
                accs = wmma_bf16(qf[dk], bf, accs);
            }
            st[nt] = accs * 0.125f;      // / sqrt(64)
        }

        // Online softmax. Lane holds rows {v + 8*hi}, col = ln; xor masks 1..8
        // reduce across the 16-lane halves independently.
#pragma unroll
        for (int v = 0; v < 8; v++) {
            float mx = fmaxf(fmaxf(st[0][v], st[1][v]), fmaxf(st[2][v], st[3][v]));
            for (int mask = 1; mask < 16; mask <<= 1)
                mx = fmaxf(mx, __shfl_xor(mx, mask, 32));
            float mn = fmaxf(m[v], mx);
            float corr = __expf(m[v] - mn);
            m[v] = mn;
            l[v] *= corr;
#pragma unroll
            for (int t = 0; t < 4; t++) o[t][v] *= corr;
            float rs = 0.f;
            float pv[4];
#pragma unroll
            for (int nt = 0; nt < 4; nt++) {
                float p = __expf(st[nt][v] - mn);
                pv[nt] = p; rs += p;
            }
            for (int mask = 1; mask < 16; mask <<= 1)
                rs += __shfl_xor(rs, mask, 32);
            l[v] += rs;
            int prow = (wave * 16 + v + hi * 8) * 64;
#pragma unroll
            for (int nt = 0; nt < 4; nt++)
                sP[prow + nt * 16 + ln] = f2bf(pv[nt]);
        }

        // O += P V  (K-dim = 64 keys -> 2 WMMA steps; N = d, 4 subtiles)
#pragma unroll
        for (int dt = 0; dt < 4; dt++) {
#pragma unroll
            for (int kk = 0; kk < 2; kk++) {
                BfFrag pa, vb;
                const unsigned short* parow = sP + (wave * 16 + ln) * 64 + kk * 32;
#pragma unroll
                for (int v = 0; v < 8; v++)
                    pa.w[v] = *(const unsigned int*)(parow + AK0(v, hi));
                const unsigned short* vrow = sVb + (dt * 16 + ln) * 64 + kk * 32 + hi * 16;
#pragma unroll
                for (int v = 0; v < 8; v++)
                    vb.w[v] = *(const unsigned int*)(vrow + 2 * v);
                o[dt] = wmma_bf16(pa, vb, o[dt]);
            }
        }
        __syncthreads();                 // all waves done with buf before overwrite
    }

    // Normalize and de-scramble into (n, hw, c) bf16.
    const int nn = nh >> 2, head = nh & 3;
#pragma unroll
    for (int dt = 0; dt < 4; dt++) {
#pragma unroll
        for (int v = 0; v < 8; v++) {
            int rr = q0 + v + hi * 8;             // head-row
            int d  = dt * 16 + ln;
            float val = o[dt][v] / l[v];
            int rowp = head * 1024 + (rr >> 2);
            int colp = ((rr & 3) << 6) + d;
            obuf[((size_t)nn * HW + rowp) * CCH + colp] = f2bf(val);
        }
    }
}

// ---------------------------------------------------------------------------
// K5: projection GEMM (16384 x 256 x 256) + bias, output transposed (n,c,h,w)
// ---------------------------------------------------------------------------
__global__ void __launch_bounds__(256)
proj_gemm(const unsigned short* __restrict__ obuf, const unsigned short* __restrict__ pwt,
          const float* __restrict__ bias, float* __restrict__ out) {
    const int lane = threadIdx.x & 31;
    const int wave = threadIdx.x >> 5;
    const int hi = (lane >> 4) & 1;
    const int ln = lane & 15;
    const int m0 = blockIdx.x * 64 + 16 * (wave & 3);
    const int c0 = blockIdx.y * 64 + 32 * (wave >> 2);

    v8f acc[2] = {vzero(), vzero()};
    for (int ks = 0; ks < 8; ks++) {
        const int kb = ks * 32;
        BfFrag a, b0, b1;
        const unsigned short* arow = obuf + (size_t)(m0 + ln) * CCH + kb;
#pragma unroll
        for (int v = 0; v < 8; v++)
            a.w[v] = *(const unsigned int*)(arow + AK0(v, hi));
        const unsigned short* b0row = pwt + (size_t)(c0 + ln) * CCH + kb + hi * 16;
        const unsigned short* b1row = b0row + 16 * CCH;
#pragma unroll
        for (int v = 0; v < 8; v++) {
            b0.w[v] = *(const unsigned int*)(b0row + 2 * v);
            b1.w[v] = *(const unsigned int*)(b1row + 2 * v);
        }
        acc[0] = wmma_bf16(a, b0, acc[0]);
        acc[1] = wmma_bf16(a, b1, acc[1]);
    }
#pragma unroll
    for (int sub = 0; sub < 2; sub++) {
#pragma unroll
        for (int v = 0; v < 8; v++) {
            int row = m0 + v + hi * 8;           // 0..16383
            int ch  = c0 + sub * 16 + ln;        // 0..255
            int nn = row >> 12, p = row & (HW - 1);
            out[((size_t)nn * CCH + ch) * HW + p] = acc[sub][v] + bias[ch];
        }
    }
}

// ---------------------------------------------------------------------------
// Host launcher
// ---------------------------------------------------------------------------
extern "C" void kernel_launch(void* const* d_in, const int* in_sizes, int n_in,
                              void* d_out, int out_size, void* d_ws, size_t ws_size,
                              hipStream_t stream) {
    const float* x      = (const float*)d_in[0];
    const float* gamma  = (const float*)d_in[1];
    const float* beta   = (const float*)d_in[2];
    const float* qkv_w  = (const float*)d_in[3];
    const float* proj_w = (const float*)d_in[4];
    const float* proj_b = (const float*)d_in[5];
    float* out = (float*)d_out;

    char* w = (char*)d_ws;
    float* scale = (float*)w;                 w += 1024;
    float* shift = (float*)w;                 w += 1024;
    unsigned short* qkv_wt  = (unsigned short*)w; w += (size_t)NQKV * CCH * 2;
    unsigned short* proj_wt = (unsigned short*)w; w += (size_t)CCH * CCH * 2;
    unsigned short* seq  = (unsigned short*)w;    w += (size_t)MROWS * CCH * 2;
    unsigned short* Qb   = (unsigned short*)w;    w += (size_t)NB * HEADS * HW * DH * 2;
    unsigned short* Kb   = (unsigned short*)w;    w += (size_t)NB * HEADS * HW * DH * 2;
    unsigned short* Vt   = (unsigned short*)w;    w += (size_t)NB * HEADS * DH * HW * 2;
    unsigned short* obuf = (unsigned short*)w;    w += (size_t)MROWS * CCH * 2;

    bn_stats<<<dim3(CCH), dim3(256), 0, stream>>>(x, gamma, beta, scale, shift);
    prep_weights<<<dim3((NQKV * CCH + CCH * CCH) / 256), dim3(256), 0, stream>>>(
        qkv_w, proj_w, qkv_wt, proj_wt);
    build_seq<<<dim3(MROWS * CCH / 256), dim3(256), 0, stream>>>(x, scale, shift, seq);
    qkv_gemm<<<dim3(MROWS / 64, NQKV / 64), dim3(256), 0, stream>>>(seq, qkv_wt, Qb, Kb, Vt);
    flash_attn<<<dim3(HW / 128, NB * HEADS), dim3(256), 0, stream>>>(Qb, Kb, Vt, obuf);
    proj_gemm<<<dim3(MROWS / 64, CCH / 64), dim3(256), 0, stream>>>(obuf, proj_wt, proj_b, out);
}
